// ActionSTGCN_83210696393174
// MI455X (gfx1250) — compile-verified, matrix operand order checked by reading, not run on hardware
//
#include <hip/hip_runtime.h>
#include <stdint.h>

#define EPSC 1e-5f
#define Bn 64
#define Tn 256
#define Vn 33
#define NCn 100
#define USE_TDM 1

typedef __attribute__((ext_vector_type(16))) __bf16 v16bf;
typedef __attribute__((ext_vector_type(8)))  float  v8f;
typedef __attribute__((ext_vector_type(4)))  unsigned int v4u;
typedef __attribute__((ext_vector_type(8)))  unsigned int v8u;

union Frag16 { v16bf v; v4u q[2]; };

__device__ __forceinline__ float bf2f(unsigned short h) {
  union { unsigned u; float f; } x; x.u = ((unsigned)h) << 16; return x.f;
}
__device__ __forceinline__ unsigned short f2bf(float f) {
  union { float f; unsigned u; } x; x.f = f;
  unsigned u = x.u;
  u += 0x7FFFu + ((u >> 16) & 1u);
  return (unsigned short)(u >> 16);
}

// ---------------- prep kernels ----------------

__global__ void prep_graph_kernel(const float* __restrict__ A, float* wconst, float* pooled) {
  int tid = threadIdx.x;
  for (int i = tid; i < Vn * Vn; i += 256) wconst[i] = A[i];
  for (int v = tid; v < Vn; v += 256) {
    float s = 0.f;
    for (int w = 0; w < Vn; ++w) s += A[v * Vn + w];
    wconst[1092 + v] = s;
  }
  for (int i = tid; i < Bn * 64; i += 256) pooled[i] = 0.f;
}

// per-block folded constants: [s1|sh1|pb|s2|sh2tot] each 64 floats.
__global__ void prep_block_consts(float* cb,
    const float* g1, const float* b1, const float* m1, const float* v1,
    const float* pb,
    const float* g2, const float* b2, const float* m2, const float* v2,
    const float* tb, const float* rb, int cout) {
  int o = threadIdx.x;
  if (o >= 64) return;
  float s1 = 0.f, sh1 = 0.f, p = 0.f, s2 = 0.f, sh2 = 0.f;
  if (o < cout) {
    s1 = g1[o] * rsqrtf(v1[o] + EPSC);
    sh1 = b1[o] - m1[o] * s1;
    p = pb[o];
    s2 = g2[o] * rsqrtf(v2[o] + EPSC);
    sh2 = b2[o] - m2[o] * s2 + tb[o] * s2;
    if (rb) sh2 += rb[o];
  }
  cb[o] = s1; cb[64 + o] = sh1; cb[128 + o] = p; cb[192 + o] = s2; cb[256 + o] = sh2;
}

// Pack weights into per-lane WMMA A-fragment layout (16x32 bf16, wave32).
__global__ void pack_kernel(unsigned short* dst, const float* __restrict__ src,
                            int cout, int cinreal, int mtiles, int ksteps, int taps, int tcnC) {
  int total = taps * mtiles * ksteps * 512;
  for (int idx = blockIdx.x * blockDim.x + threadIdx.x; idx < total; idx += gridDim.x * blockDim.x) {
    int frag = idx >> 9;
    int r = idx & 511;
    int l = r >> 4;
    int j = r & 15;
    int kstep = frag % ksteps;
    int mt = (frag / ksteps) % mtiles;
    int tap = frag / (ksteps * mtiles);
    int M = mt * 16 + (l & 15);
    int kl = (j < 8) ? (((l < 16) ? 0 : 8) + j) : (((l < 16) ? 16 : 24) + (j - 8));
    int K = kstep * 32 + kl;
    float val = 0.f;
    if (M < cout && K < cinreal)
      val = tcnC ? src[(size_t)M * tcnC * 9 + (size_t)K * 9 + tap]
                 : src[(size_t)M * cinreal + K];
    dst[idx] = f2bf(val);
  }
}

// input BN + layout change (B,C,T,V) f32 -> (B,V,T,32) bf16, C padded 6->32
__global__ void bn_in_kernel(const float* __restrict__ x,
                             const float* __restrict__ g, const float* __restrict__ bb,
                             const float* __restrict__ m, const float* __restrict__ vv,
                             unsigned short* __restrict__ out) {
  int id = blockIdx.x * blockDim.x + threadIdx.x;
  if (id >= Bn * Tn * Vn) return;
  int v = id % Vn;
  int t = (id / Vn) % Tn;
  int b = id / (Vn * Tn);
  unsigned short h[6];
#pragma unroll
  for (int c = 0; c < 6; ++c) {
    int cv = c * Vn + v;
    float inv = g[cv] * rsqrtf(vv[cv] + EPSC);
    float val = x[(((size_t)b * 6 + c) * Tn + t) * Vn + v] * inv + (bb[cv] - m[cv] * inv);
    h[c] = f2bf(val);
  }
  v4u q0, zq;
  q0[0] = (unsigned)h[0] | ((unsigned)h[1] << 16);
  q0[1] = (unsigned)h[2] | ((unsigned)h[3] << 16);
  q0[2] = (unsigned)h[4] | ((unsigned)h[5] << 16);
  q0[3] = 0u;
  zq[0] = zq[1] = zq[2] = zq[3] = 0u;
  v4u* dp = (v4u*)(out + (((size_t)b * Vn + v) * Tn + t) * 32);
  dp[0] = q0; dp[1] = zq; dp[2] = zq; dp[3] = zq;
}

// ---------------- spatial GCN: TDM stage -> A-mix (VALU) -> proj (WMMA) + BN1 + ReLU ----------
// grid: (T/16, B), 256 threads = 8 waves; wave handles joints v = wv, wv+8, ...
template <int CIN, int COUT>
__global__ void gcn_kernel(const unsigned short* __restrict__ in, unsigned short* __restrict__ g,
                           const unsigned short* __restrict__ wproj, const float* __restrict__ cb,
                           const float* __restrict__ Ag, const float* __restrict__ rsum) {
  extern __shared__ char smem[];
  unsigned short* in_s = (unsigned short*)smem;              // [V][16][CIN]
  unsigned short* m_s = in_s + Vn * 16 * CIN;                // [V][16][CIN]
  float* A_s = (float*)(m_s + Vn * 16 * CIN);                // [V][34]

  int tid = threadIdx.x;
  int t0 = blockIdx.x * 16;
  int b = blockIdx.y;

  for (int i = tid; i < Vn * Vn; i += 256) A_s[(i / Vn) * 34 + (i % Vn)] = Ag[i];

#if USE_TDM
  // One 2D TDM descriptor loads the whole [V][16*CIN] tile: per joint v a contiguous
  // row of 16*CIN halfwords at stride Tn*CIN, appended contiguously into LDS.
  if (tid < 32) {
    unsigned long long ga =
        (unsigned long long)(uintptr_t)(in + (((size_t)b * Vn) * Tn + t0) * CIN);
    unsigned lds_off = (unsigned)(uintptr_t)(void*)in_s;  // low 32 bits = LDS byte offset
    v4u d0;
    d0[0] = 1u;                                            // count=1, user descriptor
    d0[1] = lds_off;                                       // lds_addr
    d0[2] = (unsigned)ga;                                  // global_addr[31:0]
    d0[3] = ((unsigned)(ga >> 32) & 0x01FFFFFFu) | (2u << 30);  // ga[56:32] | type=2
    v8u d1;
    d1[0] = 0x00010000u;                                   // data_size=2B, no flags
    d1[1] = ((unsigned)(16 * CIN) & 0xFFFFu) << 16;        // tensor_dim0[15:0]
    d1[2] = ((unsigned)(16 * CIN) >> 16) | (33u << 16);    // tensor_dim0[31:16]|tensor_dim1[15:0]
    d1[3] = ((unsigned)(16 * CIN)) << 16;                  // tensor_dim1[31:16]|tile_dim0
    d1[4] = 33u;                                           // tile_dim1 | tile_dim2=0
    d1[5] = (unsigned)(Tn * CIN);                          // tensor_dim0_stride[31:0]
    d1[6] = 0u;                                            // stride0[47:32] | stride1 lo
    d1[7] = 0u;
    asm volatile("tensor_load_to_lds %0, %1" :: "s"(d0), "s"(d1) : "memory");
    __builtin_amdgcn_s_wait_tensorcnt(0);
  }
#else
  {
    const int CH8 = (16 * CIN) / 8;
    for (int c = tid; c < Vn * CH8; c += 256) {
      int v = c / CH8, r = c % CH8;
      const v4u* sp = (const v4u*)(in + (((size_t)b * Vn + v) * Tn + t0) * CIN);
      ((v4u*)(in_s + v * 16 * CIN))[r] = sp[r];
    }
  }
#endif
  __syncthreads();

  // adjacency mix: m[v][t][i] = sum_w A[v][w] * in[w][t][i]
  for (int e = tid; e < Vn * 16 * CIN; e += 256) {
    int v = e / (16 * CIN), r = e % (16 * CIN);
    const float* arow = A_s + v * 34;
    float acc = 0.f;
    for (int w = 0; w < Vn; ++w) acc += arow[w] * bf2f(in_s[w * 16 * CIN + r]);
    m_s[e] = f2bf(acc);
  }
  __syncthreads();

  constexpr int KST = CIN / 32;
  constexpr int MT = COUT / 16;
  int wv = tid >> 5, lane = tid & 31;
  int tl = lane & 15, hi = lane >> 4;

  for (int v = wv; v < Vn; v += 8) {
    Frag16 bf[KST];
#pragma unroll
    for (int k = 0; k < KST; ++k) {
      const unsigned short* bp = m_s + ((size_t)v * 16 + tl) * CIN + k * 32 + 8 * hi;
      bf[k].q[0] = *(const v4u*)bp;
      bf[k].q[1] = *(const v4u*)(bp + 16);
    }
    float rsv = rsum[v];
#pragma unroll
    for (int mt = 0; mt < MT; ++mt) {
      v8f acc = {0.f, 0.f, 0.f, 0.f, 0.f, 0.f, 0.f, 0.f};
#pragma unroll
      for (int k = 0; k < KST; ++k) {
        Frag16 af;
        const v4u* ap = (const v4u*)(wproj + ((size_t)(mt * KST + k)) * 512 + lane * 16);
        af.q[0] = ap[0]; af.q[1] = ap[1];
        acc = __builtin_amdgcn_wmma_f32_16x16x32_bf16(false, af.v, false, bf[k].v,
                                                      (short)0, acc, false, false);
      }
      int o0 = mt * 16 + 8 * hi;
      float vals[8];
#pragma unroll
      for (int r = 0; r < 8; ++r) {
        int o = o0 + r;
        float val = cb[o] * (acc[r] + cb[128 + o] * rsv) + cb[64 + o];
        vals[r] = val > 0.f ? val : 0.f;
      }
      v4u pk;
#pragma unroll
      for (int i = 0; i < 4; ++i)
        pk[i] = (unsigned)f2bf(vals[2 * i]) | ((unsigned)f2bf(vals[2 * i + 1]) << 16);
      *(v4u*)(g + (((size_t)b * Vn + v) * Tn + t0 + tl) * COUT + o0) = pk;
    }
  }
}

// ---------------- temporal conv: LDS-staged implicit GEMM + residual + BN2 + ReLU -------------
// grid: (T/16, V, B); blockDim = (C/16)*32, wave = one M-tile of 16 output channels.
// 24-row halo tile staged once in LDS (zeros materialized for out-of-range t),
// all waves' B-fragments come from ds_load_b128 -> 1x global traffic instead of MT*9x.
template <int C, int CRES, bool HASRES, bool FINAL>
__global__ void tcn_kernel(const unsigned short* __restrict__ g, const unsigned short* __restrict__ xin,
                           unsigned short* __restrict__ out, float* pooled,
                           const unsigned short* __restrict__ wtcn,
                           const unsigned short* __restrict__ wres,
                           const float* __restrict__ cb) {
  constexpr int KST = C / 32;
  constexpr int MT = C / 16;
  constexpr int RKST = CRES / 32;
  constexpr int NTHR = MT * 32;
  __shared__ unsigned short gs[24 * C];
  __shared__ unsigned short xs[HASRES ? 16 * CRES : 16];

  int tid = threadIdx.x;
  int lane = tid & 31, wv = tid >> 5;   // wv = M-tile
  int tl = lane & 15, hi = lane >> 4;
  int t0 = blockIdx.x * 16;
  int v = blockIdx.y;
  int b = blockIdx.z;
  const unsigned short* gbase = g + (((size_t)b * Vn + v) * Tn) * C;
  __builtin_prefetch(gbase + (size_t)(t0 + 16) * C, 0, 1);  // next t-tile

  v4u zq; zq[0] = zq[1] = zq[2] = zq[3] = 0u;

  // stage rows t0-4 .. t0+19 (zero-padded at the T edges)
  constexpr int CH = C / 8;
  for (int c = tid; c < 24 * CH; c += NTHR) {
    int row = c / CH, e = c % CH;
    int tt = t0 + row - 4;
    v4u val = zq;
    if (tt >= 0 && tt < Tn) val = ((const v4u*)(gbase + (size_t)tt * C))[e];
    ((v4u*)(gs + row * C))[e] = val;
  }
  if (HASRES) {
    const unsigned short* xb = xin + (((size_t)b * Vn + v) * Tn + t0) * CRES;
    for (int c = tid; c < 16 * (CRES / 8); c += NTHR)
      ((v4u*)xs)[c] = ((const v4u*)xb)[c];
  }
  __syncthreads();

  v8f acc = {0.f, 0.f, 0.f, 0.f, 0.f, 0.f, 0.f, 0.f};
  for (int tap = 0; tap < 9; ++tap) {
#pragma unroll
    for (int k = 0; k < KST; ++k) {
      Frag16 bf;
      const unsigned short* bp = gs + (tap + tl) * C + k * 32 + 8 * hi;
      bf.q[0] = *(const v4u*)bp;
      bf.q[1] = *(const v4u*)(bp + 16);
      Frag16 af;
      const v4u* ap = (const v4u*)(wtcn + ((size_t)((tap * MT + wv) * KST + k)) * 512 + lane * 16);
      af.q[0] = ap[0]; af.q[1] = ap[1];
      acc = __builtin_amdgcn_wmma_f32_16x16x32_bf16(false, af.v, false, bf.v,
                                                    (short)0, acc, false, false);
    }
  }

  int o0 = wv * 16 + 8 * hi;
  float radd[8];
  if (HASRES) {
    v8f rac = {0.f, 0.f, 0.f, 0.f, 0.f, 0.f, 0.f, 0.f};
#pragma unroll
    for (int k = 0; k < RKST; ++k) {
      Frag16 bf;
      const unsigned short* bp = xs + tl * CRES + k * 32 + 8 * hi;
      bf.q[0] = *(const v4u*)bp;
      bf.q[1] = *(const v4u*)(bp + 16);
      Frag16 af;
      const v4u* ap = (const v4u*)(wres + ((size_t)(wv * RKST + k)) * 512 + lane * 16);
      af.q[0] = ap[0]; af.q[1] = ap[1];
      rac = __builtin_amdgcn_wmma_f32_16x16x32_bf16(false, af.v, false, bf.v,
                                                    (short)0, rac, false, false);
    }
#pragma unroll
    for (int r = 0; r < 8; ++r) radd[r] = rac[r];
  } else {
    v4u q = *(const v4u*)(xin + (((size_t)b * Vn + v) * Tn + t0 + tl) * C + o0);
#pragma unroll
    for (int i = 0; i < 4; ++i) {
      radd[2 * i] = bf2f((unsigned short)(q[i] & 0xFFFFu));
      radd[2 * i + 1] = bf2f((unsigned short)(q[i] >> 16));
    }
  }

  if (!FINAL) {
    float vals[8];
#pragma unroll
    for (int r = 0; r < 8; ++r) {
      int o = o0 + r;
      float val = cb[192 + o] * acc[r] + cb[256 + o] + radd[r];
      vals[r] = val > 0.f ? val : 0.f;
    }
    v4u pk;
#pragma unroll
    for (int i = 0; i < 4; ++i)
      pk[i] = (unsigned)f2bf(vals[2 * i]) | ((unsigned)f2bf(vals[2 * i + 1]) << 16);
    *(v4u*)(out + (((size_t)b * Vn + v) * Tn + t0 + tl) * C + o0) = pk;
  } else {
#pragma unroll
    for (int r = 0; r < 8; ++r) {
      int o = o0 + r;
      float val = cb[192 + o] * acc[r] + cb[256 + o] + radd[r];
      val = val > 0.f ? val : 0.f;
      for (int mbit = 1; mbit < 16; mbit <<= 1) val += __shfl_xor(val, mbit, 32);
      if (tl == 0) atomicAdd(&pooled[b * 64 + o], val);
    }
  }
}

__global__ void fc_kernel(const float* __restrict__ pooled, const float* __restrict__ fw,
                          const float* __restrict__ fb, float* __restrict__ outp) {
  int b = blockIdx.x;
  int nc = threadIdx.x;
  if (nc >= NCn) return;
  const float scale = 1.0f / (float)(Tn * Vn);
  float acc = fb[nc];
  for (int o = 0; o < 64; ++o) acc += pooled[b * 64 + o] * scale * fw[nc * 64 + o];
  outp[b * NCn + nc] = acc;
}

// ---------------- launch ----------------

extern "C" void kernel_launch(void* const* d_in, const int* in_sizes, int n_in,
                              void* d_out, int out_size, void* d_ws, size_t ws_size,
                              hipStream_t stream) {
  (void)in_sizes; (void)n_in; (void)out_size; (void)ws_size;
  const float* x = (const float*)d_in[0];
  const float* Amat = (const float*)d_in[1];

  char* ws = (char*)d_ws;
  const size_t SZ32B = (size_t)Bn * Vn * Tn * 32 * 2;  // 34,603,008 B
  unsigned short* in0 = (unsigned short*)(ws);
  unsigned short* g0  = (unsigned short*)(ws + SZ32B);
  unsigned short* h0  = (unsigned short*)(ws + 2 * SZ32B);
  unsigned short* g1  = (unsigned short*)(ws);               // reuse [0, 2*SZ32B)
  unsigned short* h1  = (unsigned short*)(ws + 3 * SZ32B);
  unsigned short* g2  = (unsigned short*)(ws);               // reuse again
  float* pooled = (float*)(ws + 5 * SZ32B);
  char* wb = ws + 5 * SZ32B + 65536;                          // packed weights (~174 MB total ws)
  unsigned short* W_PROJ0 = (unsigned short*)(wb + 0 * 131072);
  unsigned short* W_PROJ1 = (unsigned short*)(wb + 1 * 131072);
  unsigned short* W_PROJ2 = (unsigned short*)(wb + 2 * 131072);
  unsigned short* W_TCN0  = (unsigned short*)(wb + 3 * 131072);
  unsigned short* W_TCN1  = (unsigned short*)(wb + 4 * 131072);
  unsigned short* W_TCN2  = (unsigned short*)(wb + 5 * 131072);
  unsigned short* W_RES0  = (unsigned short*)(wb + 6 * 131072);
  unsigned short* W_RES1  = (unsigned short*)(wb + 7 * 131072);
  float* wconst = (float*)(wb + 8 * 131072);
  float* rsum = wconst + 1092;
  float* cb0 = wconst + 1152;
  float* cb1 = cb0 + 320;
  float* cb2 = cb1 + 320;

  prep_graph_kernel<<<1, 256, 0, stream>>>(Amat, wconst, pooled);
  prep_block_consts<<<1, 64, 0, stream>>>(cb0,
      (const float*)d_in[8], (const float*)d_in[9], (const float*)d_in[10], (const float*)d_in[11],
      (const float*)d_in[7],
      (const float*)d_in[14], (const float*)d_in[15], (const float*)d_in[16], (const float*)d_in[17],
      (const float*)d_in[13], (const float*)d_in[19], 32);
  prep_block_consts<<<1, 64, 0, stream>>>(cb1,
      (const float*)d_in[22], (const float*)d_in[23], (const float*)d_in[24], (const float*)d_in[25],
      (const float*)d_in[21],
      (const float*)d_in[28], (const float*)d_in[29], (const float*)d_in[30], (const float*)d_in[31],
      (const float*)d_in[27], (const float*)d_in[33], 64);
  prep_block_consts<<<1, 64, 0, stream>>>(cb2,
      (const float*)d_in[36], (const float*)d_in[37], (const float*)d_in[38], (const float*)d_in[39],
      (const float*)d_in[35],
      (const float*)d_in[42], (const float*)d_in[43], (const float*)d_in[44], (const float*)d_in[45],
      (const float*)d_in[41], (const float*)nullptr, 64);

  pack_kernel<<<8, 256, 0, stream>>>(W_PROJ0, (const float*)d_in[6], 32, 6, 2, 1, 1, 0);
  pack_kernel<<<8, 256, 0, stream>>>(W_PROJ1, (const float*)d_in[20], 64, 32, 4, 1, 1, 0);
  pack_kernel<<<16, 256, 0, stream>>>(W_PROJ2, (const float*)d_in[34], 64, 64, 4, 2, 1, 0);
  pack_kernel<<<36, 256, 0, stream>>>(W_TCN0, (const float*)d_in[12], 32, 32, 2, 1, 9, 32);
  pack_kernel<<<144, 256, 0, stream>>>(W_TCN1, (const float*)d_in[26], 64, 64, 4, 2, 9, 64);
  pack_kernel<<<144, 256, 0, stream>>>(W_TCN2, (const float*)d_in[40], 64, 64, 4, 2, 9, 64);
  pack_kernel<<<8, 256, 0, stream>>>(W_RES0, (const float*)d_in[18], 32, 6, 2, 1, 1, 0);
  pack_kernel<<<8, 256, 0, stream>>>(W_RES1, (const float*)d_in[32], 64, 32, 4, 1, 1, 0);

  bn_in_kernel<<<(Bn * Tn * Vn + 255) / 256, 256, 0, stream>>>(
      x, (const float*)d_in[2], (const float*)d_in[3], (const float*)d_in[4], (const float*)d_in[5], in0);

  size_t sh32 = 2 * (size_t)Vn * 16 * 32 * 2 + (size_t)Vn * 34 * 4;  // ~72 KB
  size_t sh64 = 2 * (size_t)Vn * 16 * 64 * 2 + (size_t)Vn * 34 * 4;  // ~140 KB (<=320 KB/WG)

  gcn_kernel<32, 32><<<dim3(Tn / 16, Bn), 256, sh32, stream>>>(in0, g0, W_PROJ0, cb0, wconst, rsum);
  tcn_kernel<32, 32, true, false><<<dim3(Tn / 16, Vn, Bn), 64, 0, stream>>>(
      g0, in0, h0, nullptr, W_TCN0, W_RES0, cb0);

  gcn_kernel<32, 64><<<dim3(Tn / 16, Bn), 256, sh32, stream>>>(h0, g1, W_PROJ1, cb1, wconst, rsum);
  tcn_kernel<64, 32, true, false><<<dim3(Tn / 16, Vn, Bn), 128, 0, stream>>>(
      g1, h0, h1, nullptr, W_TCN1, W_RES1, cb1);

  gcn_kernel<64, 64><<<dim3(Tn / 16, Bn), 256, sh64, stream>>>(h1, g2, W_PROJ2, cb2, wconst, rsum);
  tcn_kernel<64, 64, false, true><<<dim3(Tn / 16, Vn, Bn), 128, 0, stream>>>(
      g2, h1, nullptr, pooled, W_TCN2, nullptr, cb2);

  fc_kernel<<<Bn, 128, 0, stream>>>(pooled, (const float*)d_in[46], (const float*)d_in[47], (float*)d_out);
}
